// MODEL_2405181686409
// MI455X (gfx1250) — matrix operand order repaired and checked by reading, not compiled, not run on hardware
//
#include <hip/hip_runtime.h>
#include <hip/hip_bf16.h>
#include <math.h>

// ---------------------------------------------------------------------------
// Model dims
// ---------------------------------------------------------------------------
#define BB   512
#define SS   16
#define DD   512
#define HH   256
#define DKK  256
#define NHH  4
#define HDD  64            // DK / NH
#define ROWS (BB*SS)       // 8192
#define UDIM 400
#define CATN 50
#define SUBN 500
#define CONN 5000

typedef __attribute__((ext_vector_type(16))) __bf16 v16bf;
typedef __attribute__((ext_vector_type(8)))  float  v8f;

__device__ __forceinline__ unsigned short bf16bits(float f) {
    union { float f; unsigned u; } x; x.f = f;
    unsigned r = x.u + 0x7fffu + ((x.u >> 16) & 1u);   // round-to-nearest-even
    return (unsigned short)(r >> 16);
}
__device__ __forceinline__ __bf16 f2bf(float f) {
    union { unsigned short s; __bf16 b; } y;
    y.s = bf16bits(f);
    return y.b;
}

__device__ __forceinline__ float sigm(float x) { return 1.0f / (1.0f + __expf(-x)); }

// CDNA5 async copy: global -> LDS, 16B per lane, tracked by ASYNCcnt.
__device__ __forceinline__ void async_b128(unsigned lds_addr, const void* gptr) {
    asm volatile("global_load_async_to_lds_b128 %0, %1, off"
                 :: "v"(lds_addr), "v"((unsigned long long)(size_t)gptr)
                 : "memory");
}
__device__ __forceinline__ void wait_async0() {
    asm volatile("s_wait_asynccnt 0x0" ::: "memory");
}
__device__ __forceinline__ void wait_async3() {   // allow 3 outstanding (next chunk)
    asm volatile("s_wait_asynccnt 0x3" ::: "memory");
}

__device__ __forceinline__ float block_reduce_sum(float v, float* red, int tid, int n) {
    red[tid] = v; __syncthreads();
    for (int off = n >> 1; off; off >>= 1) {
        if (tid < off) red[tid] += red[tid + off];
        __syncthreads();
    }
    float r = red[0]; __syncthreads();
    return r;
}

// ---------------------------------------------------------------------------
// FAST bf16 GEMM:  C[m,n] = act( sum_k A[m,k]*W[n,k] + bias[n] )
// A, W bf16 in global (converted once). C fp32 / Cbf bf16 outputs (nullable).
// Preconditions: M % 128 == 0, K % 32 == 0, lda % 8 == 0, ldw % 8 == 0,
// 32B-aligned bases. Ragged N: clamped loads + guarded stores.
// 256 threads = 8 waves; block tile 128x64; wave w owns rows [16w,16w+16) x
// all 64 cols -> 4 WMMAs per 32-wide K chunk.
// DOUBLE-BUFFERED async pipeline: chunk k+1 is prefetched with
// GLOBAL_LOAD_ASYNC_TO_LDS_B128 while chunk k computes; in-order ASYNCcnt
// completion means s_wait_asynccnt<=3 releases the current chunk.
// ---------------------------------------------------------------------------
#define ABUF 10240u   // 128 rows * 80B
#define BBUF 5120u    //  64 rows * 80B

__global__ void __launch_bounds__(256)
gemm_bf16_async(const unsigned short* __restrict__ A, int lda,
                const unsigned short* __restrict__ W, int ldw,
                const float* __restrict__ bias,
                float* __restrict__ C, int ldc,
                unsigned short* __restrict__ Cbf, int ldcb,
                int M, int N, int K, int act)
{
    __shared__ unsigned Asu[2][128][20];   // [buf][m][k/2], rows padded to 80B
    __shared__ unsigned Bsu[2][64][20];

    const int bm   = blockIdx.y * 128;
    const int bn   = blockIdx.x * 64;
    const int tid  = threadIdx.x;
    const int wave = tid >> 5;
    const int lane = tid & 31;
    const int half = lane >> 4;
    const int mr   = lane & 15;

    // fill slots: A = 128 rows x 4 segs (16B), B = 64 rows x 4 segs
    const int arow = tid >> 2;          // 0..63 (+64 for second A slot)
    const int seg  = tid & 3;
    const int brc  = (bn + arow < N) ? (bn + arow) : (N - 1);

    const unsigned asub = (unsigned)(size_t)&Asu[0][0][0];
    const unsigned bsub = (unsigned)(size_t)&Bsu[0][0][0];
    const unsigned la   = (unsigned)arow * 80u + (unsigned)seg * 16u;

    const unsigned short* gA0 = A + (size_t)(bm + arow)      * lda + seg * 8;
    const unsigned short* gA1 = A + (size_t)(bm + arow + 64) * lda + seg * 8;
    const unsigned short* gW  = W + (size_t)brc              * ldw + seg * 8;

    union { v8f v; float f[8]; } acc[4];
    #pragma unroll
    for (int j = 0; j < 4; ++j)
        #pragma unroll
        for (int i = 0; i < 8; ++i) acc[j].f[i] = 0.0f;

    // prologue: prefetch chunk 0 into buffer 0
    async_b128(asub + la,        gA0);
    async_b128(asub + la + BBUF, gA1);      // +64 rows = 5120B
    async_b128(bsub + la,        gW);

    int buf = 0;
    for (int k0 = 0; k0 < K; k0 += 32) {
        const int nk = k0 + 32;
        if (nk < K) {   // prefetch next chunk into the other buffer
            const unsigned nb = (unsigned)(buf ^ 1);
            async_b128(asub + nb * ABUF + la,        gA0 + nk);
            async_b128(asub + nb * ABUF + la + BBUF, gA1 + nk);
            async_b128(bsub + nb * BBUF + la,        gW  + nk);
            wait_async3();    // current chunk's 3 loads (in-order) have landed
        } else {
            wait_async0();
        }
        __syncthreads();

        // ---- fragments (ISA bf16 A/B register layouts) ----
        union { v16bf v; uint4 q[2]; } af, bf[4];
        const unsigned* arp = &Asu[buf][wave * 16 + mr][0];
        af.q[0] = *(const uint4*)(arp + half * 4);
        af.q[1] = *(const uint4*)(arp + 8 + half * 4);
        #pragma unroll
        for (int j = 0; j < 4; ++j) {
            const unsigned* brp = &Bsu[buf][j * 16 + mr][0];
            bf[j].q[0] = *(const uint4*)(brp + half * 4);
            bf[j].q[1] = *(const uint4*)(brp + 8 + half * 4);
        }
        #pragma unroll
        for (int j = 0; j < 4; ++j)
            acc[j].v = __builtin_amdgcn_wmma_f32_16x16x32_bf16(
                false, af.v, false, bf[j].v, (short)0, acc[j].v, false, false);
        __syncthreads();   // reads done before this buffer is refilled
        buf ^= 1;
    }

    // ---- epilogue: reg r -> M = r + 8*half ; N = (lane&15) + 16*j ----
    #pragma unroll
    for (int j = 0; j < 4; ++j) {
        const int n = bn + j * 16 + mr;
        const float bv = (bias && n < N) ? bias[n] : 0.0f;
        #pragma unroll
        for (int r = 0; r < 8; ++r) {
            int m = bm + wave * 16 + r + 8 * half;
            float v = acc[j].f[r] + bv;
            if (act == 1) v = fmaxf(v, 0.0f);
            if (n < N) {
                if (C)   C[(size_t)m * ldc + n]    = v;
                if (Cbf) Cbf[(size_t)m * ldcb + n] = bf16bits(v);
            }
        }
    }
}

// ---------------------------------------------------------------------------
// Generic guarded GEMM, fp32 inputs (only proj_u: K=400).
// ---------------------------------------------------------------------------
__global__ void gemm_wmma(const float* __restrict__ A, int lda,
                          const float* __restrict__ W, int ldw,
                          const float* __restrict__ bias,
                          float* __restrict__ C, int ldc,
                          unsigned short* __restrict__ Cbf, int ldcb,
                          int M, int N, int K, int act)
{
    __shared__ __bf16 As[32][34];
    __shared__ __bf16 Bs[32][34];

    const int bm   = blockIdx.y * 32;
    const int bn   = blockIdx.x * 32;
    const int tid  = threadIdx.x;
    const int wave = tid >> 5;
    const int lane = tid & 31;
    const int wm   = (wave >> 1) * 16;
    const int wn   = (wave & 1) * 16;
    const int half = lane >> 4;
    const int mr   = lane & 15;

    union { v8f v; float f[8]; } acc;
    #pragma unroll
    for (int i = 0; i < 8; ++i) acc.f[i] = 0.0f;

    for (int k0 = 0; k0 < K; k0 += 32) {
        for (int i = tid; i < 32 * 32; i += 128) {
            int r = i >> 5, c = i & 31;
            int gk = k0 + c;
            int gm = bm + r;
            float a = (gm < M && gk < K) ? A[(size_t)gm * lda + gk] : 0.0f;
            As[r][c] = f2bf(a);
            int gn = bn + r;
            float w = (gn < N && gk < K) ? W[(size_t)gn * ldw + gk] : 0.0f;
            Bs[r][c] = f2bf(w);
        }
        __syncthreads();

        union { v16bf v; __bf16 e[16]; } af, bfv;
        #pragma unroll
        for (int i = 0; i < 8; ++i) {
            int kb = half * 16 + 2 * i;
            af.e[2 * i]     = As[wm + mr][kb];
            af.e[2 * i + 1] = As[wm + mr][kb + 1];
        }
        #pragma unroll
        for (int i = 0; i < 8; ++i) {
            int kb = half * 16 + 2 * i;
            bfv.e[2 * i]     = Bs[wn + mr][kb];
            bfv.e[2 * i + 1] = Bs[wn + mr][kb + 1];
        }
        acc.v = __builtin_amdgcn_wmma_f32_16x16x32_bf16(
            false, af.v, false, bfv.v, (short)0, acc.v, false, false);
        __syncthreads();
    }

    const int n = bn + wn + mr;
    #pragma unroll
    for (int r = 0; r < 8; ++r) {
        int m = bm + wm + r + 8 * half;
        if (m < M && n < N) {
            float v = acc.f[r] + (bias ? bias[n] : 0.0f);
            if (act == 1) v = fmaxf(v, 0.0f);
            if (C)   C[(size_t)m * ldc + n]    = v;
            if (Cbf) Cbf[(size_t)m * ldcb + n] = bf16bits(v);
        }
    }
}

// ---------------------------------------------------------------------------
// Elementwise / glue kernels
// ---------------------------------------------------------------------------
__global__ void zero_kernel(float* p, int n) {
    int i = blockIdx.x * blockDim.x + threadIdx.x;
    if (i < n) p[i] = 0.0f;
}

__global__ void cvt_bf16_kernel(const float* __restrict__ src,
                                unsigned short* __restrict__ dst, int n) {
    int i = blockIdx.x * blockDim.x + threadIdx.x;
    if (i < n) dst[i] = bf16bits(src[i]);
}

__global__ void concat_emb_kernel(const float* __restrict__ visual,
                                  const float* __restrict__ text,
                                  unsigned short* __restrict__ embb,
                                  unsigned short* __restrict__ vtinb)
{
    int i = blockIdx.x * blockDim.x + threadIdx.x;
    if (i >= ROWS * DD) return;
    int r = i / DD, c = i % DD;
    unsigned short v = bf16bits(visual[i]), t = bf16bits(text[i]);
    embb[(size_t)r * 1024 + c]         = v;
    embb[(size_t)r * 1024 + 512 + c]   = t;
    vtinb[(size_t)r * 2048 + c]        = v;
    vtinb[(size_t)r * 2048 + 512 + c]  = t;
}

__global__ void user_embed_kernel(const float* __restrict__ user,
                                  const float* __restrict__ user_emb,
                                  unsigned short* __restrict__ vtinb)
{
    int i = blockIdx.x * blockDim.x + threadIdx.x;
    if (i >= ROWS * 256) return;
    int r = i / 256, j = i % 256;
    int uid = (int)user[(size_t)r * (UDIM + 1)];
    vtinb[(size_t)r * 2048 + 1024 + j] = bf16bits(user_emb[(size_t)uid * 256 + j]);
}

__global__ void gate1_kernel(const float* __restrict__ l, const float* __restrict__ t,
                             const float* __restrict__ catemb,
                             float* __restrict__ subf, unsigned short* __restrict__ subfb)
{
    int i = blockIdx.x * blockDim.x + threadIdx.x;
    if (i >= ROWS * 512) return;
    int r = i / 512, c = i % 512;
    float val = catemb[i] * sigm(l[i] + t[i]);
    subf[(size_t)r * 1024 + 512 + c]  = val;
    subfb[(size_t)r * 1024 + 512 + c] = bf16bits(val);
}

__global__ void gate2_kernel(const float* __restrict__ l2, const float* __restrict__ t2,
                             const float* __restrict__ subf,
                             unsigned short* __restrict__ confb)
{
    int i = blockIdx.x * blockDim.x + threadIdx.x;
    if (i >= ROWS * 1024) return;
    int r = i / 1024, c = i % 1024;
    float val = subf[i] * sigm(l2[i] + t2[(size_t)r * 512 + (c & 511)]);
    confb[(size_t)r * 1536 + 512 + c] = bf16bits(val);
}

__global__ void ce_kernel(const float* __restrict__ logits, int ldl, int ncls,
                          const int* __restrict__ category, int head, int row0,
                          float* __restrict__ acc)
{
    __shared__ float red[256];
    int r = blockIdx.x, tid = threadIdx.x;
    const float* row = logits + (size_t)r * ldl;
    float mx = -1e30f;
    for (int i = tid; i < ncls; i += 256) mx = fmaxf(mx, row[i]);
    red[tid] = mx; __syncthreads();
    for (int off = 128; off; off >>= 1) {
        if (tid < off) red[tid] = fmaxf(red[tid], red[tid + off]);
        __syncthreads();
    }
    float m = red[0]; __syncthreads();
    float s = 0.0f;
    for (int i = tid; i < ncls; i += 256) s += __expf(row[i] - m);
    red[tid] = s; __syncthreads();
    for (int off = 128; off; off >>= 1) {
        if (tid < off) red[tid] += red[tid + off];
        __syncthreads();
    }
    if (tid == 0) {
        int tgt = category[(size_t)(row0 + r) * 3 + head];
        atomicAdd(acc, m + logf(red[0]) - row[tgt]);
    }
}

// LayerNorm-LSTM cell; h -> bf16, outs -> fp32 + bf16 (GEMM operand), c fp32.
__global__ void lstm_cell_kernel(const float* __restrict__ zi, const float* __restrict__ zh,
                                 unsigned short* __restrict__ hb, float* __restrict__ c,
                                 float* __restrict__ outs_t,
                                 unsigned short* __restrict__ outsb_t,
                                 const float* __restrict__ gih, const float* __restrict__ bih,
                                 const float* __restrict__ ghh, const float* __restrict__ bhh,
                                 const float* __restrict__ gc,  const float* __restrict__ bc)
{
    __shared__ float red[256];
    __shared__ float zb[1024];
    int b = blockIdx.x, tid = threadIdx.x;
    const float* zir = zi + (size_t)b * 1024;
    const float* zhr = zh + (size_t)b * 1024;
    float vi[4], vh[4], si = 0.0f, sh = 0.0f;
    #pragma unroll
    for (int k = 0; k < 4; ++k) {
        int j = tid + 256 * k;
        vi[k] = zir[j]; vh[k] = zhr[j];
        si += vi[k];    sh += vh[k];
    }
    float mi = block_reduce_sum(si, red, tid, 256) * (1.0f / 1024.0f);
    float mh = block_reduce_sum(sh, red, tid, 256) * (1.0f / 1024.0f);
    float qi = 0.0f, qh = 0.0f;
    #pragma unroll
    for (int k = 0; k < 4; ++k) {
        float a = vi[k] - mi, d = vh[k] - mh;
        qi += a * a; qh += d * d;
    }
    float ri = rsqrtf(block_reduce_sum(qi, red, tid, 256) * (1.0f / 1024.0f) + 1e-5f);
    float rh = rsqrtf(block_reduce_sum(qh, red, tid, 256) * (1.0f / 1024.0f) + 1e-5f);
    #pragma unroll
    for (int k = 0; k < 4; ++k) {
        int j = tid + 256 * k;
        zb[j] = ((vi[k] - mi) * ri * gih[j] + bih[j]) + ((vh[k] - mh) * rh * ghh[j] + bhh[j]);
    }
    __syncthreads();
    float iv = zb[tid], fv = zb[256 + tid], gv = zb[512 + tid], ov = zb[768 + tid];
    float cn = sigm(fv) * c[(size_t)b * 256 + tid] + sigm(iv) * tanhf(gv);
    float mc = block_reduce_sum(cn, red, tid, 256) * (1.0f / 256.0f);
    float dc = cn - mc;
    float vc = block_reduce_sum(dc * dc, red, tid, 256) * (1.0f / 256.0f);
    float lnc = dc * rsqrtf(vc + 1e-5f) * gc[tid] + bc[tid];
    float hn = sigm(ov) * tanhf(lnc);
    c[(size_t)b * 256 + tid]       = cn;
    hb[(size_t)b * 256 + tid]      = bf16bits(hn);
    outs_t[(size_t)b * 256 + tid]  = hn;
    outsb_t[(size_t)b * 256 + tid] = bf16bits(hn);
}

// scores[b,h,s] = dot64(K[b,s,h*64:], q[b,h*64:]) / 8   (K precomputed by GEMM)
__global__ void attn_score_kernel(const float* __restrict__ kb, const float* __restrict__ q,
                                  float* __restrict__ scores)
{
    __shared__ float red[256];
    int row = blockIdx.x;           // b*16 + s
    int b = row >> 4, s = row & 15;
    int tid = threadIdx.x;
    float kj = kb[((size_t)s * BB + b) * 256 + tid];
    red[tid] = kj * q[(size_t)b * 256 + tid];
    __syncthreads();
    for (int off = 32; off >= 1; off >>= 1) {
        if ((tid & 63) < off) red[tid] += red[tid + off];
        __syncthreads();
    }
    if ((tid & 63) == 0)
        scores[((size_t)b * NHH + (tid >> 6)) * SS + s] = red[tid] * 0.125f;
}

__global__ void attn_softmax_kernel(float* __restrict__ scores)
{
    __shared__ float buf[SS];
    int bh = blockIdx.x, tid = threadIdx.x;
    float* w = scores + (size_t)bh * SS;
    if (tid < SS) buf[tid] = w[tid];
    __syncthreads();
    if (tid == 0) {
        float mx = buf[0];
        for (int i = 1; i < SS; ++i) mx = fmaxf(mx, buf[i]);
        float s = 0.0f;
        for (int i = 0; i < SS; ++i) { buf[i] = __expf(buf[i] - mx); s += buf[i]; }
        float inv = 1.0f / s;
        for (int i = 0; i < SS; ++i) w[i] = buf[i] * inv;
    }
}

// ATTbf[b, co] = bv[co] + sum_s sw[b,h,s] * VAL[(s,b), co]  (softmax sums to 1)
__global__ void attn_reduce_kernel(const float* __restrict__ val, const float* __restrict__ sw,
                                   const float* __restrict__ bv,
                                   unsigned short* __restrict__ attb)
{
    int i = blockIdx.x * blockDim.x + threadIdx.x;
    if (i >= BB * NHH * DKK) return;
    int b = i >> 10, co = i & 1023;        // co = h*256 + o
    int hh = co >> 8;
    const float* swr = sw + ((size_t)b * NHH + hh) * SS;
    float acc = bv[co];
    #pragma unroll
    for (int s = 0; s < SS; ++s)
        acc += swr[s] * val[((size_t)s * BB + b) * 1024 + co];
    attb[(size_t)b * 1024 + co] = bf16bits(acc);
}

__global__ void final_out_kernel(const float* __restrict__ att2, const float* __restrict__ Wo,
                                 const float* __restrict__ bo, const float* __restrict__ label,
                                 float* __restrict__ out, float* __restrict__ mseacc)
{
    __shared__ float red[256];
    int b = blockIdx.x, tid = threadIdx.x;
    red[tid] = att2[(size_t)b * 256 + tid] * Wo[tid];
    __syncthreads();
    for (int off = 128; off; off >>= 1) {
        if (tid < off) red[tid] += red[tid + off];
        __syncthreads();
    }
    if (tid == 0) {
        float o = red[0] + bo[0];
        out[b] = o;
        float d = o - label[(size_t)b * SS + (SS - 1)];
        atomicAdd(mseacc, d * d);
    }
}

__global__ void loss_kernel(const float* __restrict__ acc, float* __restrict__ out0)
{
    out0[0] = acc[3] * (1.0f / BB)
            + 0.01f * ((acc[0] + acc[1] + acc[2]) * (1.0f / (float)ROWS));
}

// ---------------------------------------------------------------------------
// Host orchestration
// ---------------------------------------------------------------------------
static inline void gemmb(hipStream_t s, const unsigned short* A, int lda,
                         const unsigned short* W, int ldw, const float* bias,
                         float* C, int ldc, unsigned short* Cbf, int ldcb,
                         int M, int N, int K, int act)
{
    dim3 g((N + 63) / 64, M / 128), b(256);
    gemm_bf16_async<<<g, b, 0, s>>>(A, lda, W, ldw, bias, C, ldc, Cbf, ldcb, M, N, K, act);
}

// workspace offsets (floats)
#define OFF_CATE   0ULL
#define OFF_SUBF   4194304ULL
#define OFF_SCR    12582912ULL
#define OFF_EMBB   25165824ULL
#define OFF_CATEB  29360128ULL
#define OFF_SUBFB  31457280ULL
#define OFF_CONFB  35651584ULL
#define OFF_VTINB  41943040ULL
#define OFF_VTB    50331648ULL
#define OFF_WBF    52428800ULL
#define OFF_HB     59768832ULL
#define OFF_C      59834368ULL
#define OFF_ZI     59965440ULL
#define OFF_ZH     60489728ULL
#define OFF_OUTS   61014016ULL
#define OFF_Q      63111168ULL
#define OFF_SCORES 63242240ULL
#define OFF_ATTB   63275008ULL
#define OFF_ATT2   63537152ULL
#define OFF_ACC    63668224ULL
#define OFF_OUTSB  63668240ULL     //  2,097,152 bf16 (1,048,576 f)
#define OFF_VAL    64716816ULL     //  8,388,608 fp32
#define OFF_KB     73105424ULL     //  2,097,152 fp32
#define OFF_WKB    75202576ULL     //     65,536 bf16
#define OFF_WVB    75235344ULL     //    262,144 bf16

// bf16 weight sub-offsets (bf16 elements within WBF)
#define WB_EMB_CAT 0ULL
#define WB_EMB_SUB 524288ULL
#define WB_EMB_CON 1048576ULL
#define WB_F1_THIS 1572864ULL
#define WB_F1_LAST 1835008ULL
#define WB_F2_LAST 2097152ULL
#define WB_F2_THIS 3145728ULL
#define WB_OUT_CAT 3407872ULL
#define WB_OUT_SUB 3433472ULL
#define WB_OUT_CON 3945472ULL
#define WB_FUSE    11625472ULL
#define WB_MLP     12411904ULL
#define WB_WIH     13460480ULL
#define WB_WHH     13984768ULL
#define WB_WQ      14246912ULL
#define WB_WOUT    14312448ULL

extern "C" void kernel_launch(void* const* d_in, const int* in_sizes, int n_in,
                              void* d_out, int out_size, void* d_ws, size_t ws_size,
                              hipStream_t stream)
{
    (void)in_sizes; (void)n_in; (void)out_size; (void)ws_size;
    const float* visual   = (const float*)d_in[0];
    const float* text     = (const float*)d_in[1];
    const int*   category = (const int*)  d_in[2];
    const float* user     = (const float*)d_in[3];
    const float* label    = (const float*)d_in[4];
    const float* h0       = (const float*)d_in[5];
    const float* c0       = (const float*)d_in[6];
    const float* user_emb   = (const float*)d_in[7];
    const float* proj_u_W   = (const float*)d_in[8];
    const float* proj_u_b   = (const float*)d_in[9];
    const float* emb_cat_W  = (const float*)d_in[10];
    const float* emb_cat_b  = (const float*)d_in[11];
    const float* emb_sub_W  = (const float*)d_in[12];
    const float* emb_sub_b  = (const float*)d_in[13];
    const float* emb_con_W  = (const float*)d_in[14];
    const float* emb_con_b  = (const float*)d_in[15];
    const float* f1_last_W  = (const float*)d_in[16];
    const float* f1_last_b  = (const float*)d_in[17];
    const float* f1_this_W  = (const float*)d_in[18];
    const float* f1_this_b  = (const float*)d_in[19];
    const float* f2_last_W  = (const float*)d_in[20];
    const float* f2_last_b  = (const float*)d_in[21];
    const float* f2_this_W  = (const float*)d_in[22];
    const float* f2_this_b  = (const float*)d_in[23];
    const float* out_cat_W  = (const float*)d_in[24];
    const float* out_cat_b  = (const float*)d_in[25];
    const float* out_sub_W  = (const float*)d_in[26];
    const float* out_sub_b  = (const float*)d_in[27];
    const float* out_con_W  = (const float*)d_in[28];
    const float* out_con_b  = (const float*)d_in[29];
    const float* fuse_W     = (const float*)d_in[30];
    const float* fuse_b     = (const float*)d_in[31];
    const float* mlp_W      = (const float*)d_in[32];
    const float* mlp_b      = (const float*)d_in[33];
    const float* W_ih       = (const float*)d_in[34];
    const float* W_hh       = (const float*)d_in[35];
    const float* b_ih       = (const float*)d_in[36];
    const float* b_hh       = (const float*)d_in[37];
    const float* ln_ih_g    = (const float*)d_in[38];
    const float* ln_ih_b    = (const float*)d_in[39];
    const float* ln_hh_g    = (const float*)d_in[40];
    const float* ln_hh_b    = (const float*)d_in[41];
    const float* ln_c_g     = (const float*)d_in[42];
    const float* ln_c_b     = (const float*)d_in[43];
    const float* Wq         = (const float*)d_in[44];
    const float* bq         = (const float*)d_in[45];
    const float* Wk         = (const float*)d_in[46];
    const float* bk         = (const float*)d_in[47];
    const float* Wv         = (const float*)d_in[48];
    const float* bv         = (const float*)d_in[49];
    const float* Wout       = (const float*)d_in[50];
    const float* bout       = (const float*)d_in[51];
    const float* Wo         = (const float*)d_in[52];
    const float* bo         = (const float*)d_in[53];

    float* ws = (float*)d_ws;
    float*          CATE  = ws + OFF_CATE;
    float*          SUBF  = ws + OFF_SUBF;
    float*          SCR   = ws + OFF_SCR;
    unsigned short* EMBB  = (unsigned short*)(ws + OFF_EMBB);
    unsigned short* CATEB = (unsigned short*)(ws + OFF_CATEB);
    unsigned short* SUBFB = (unsigned short*)(ws + OFF_SUBFB);
    unsigned short* CONFB = (unsigned short*)(ws + OFF_CONFB);
    unsigned short* VTINB = (unsigned short*)(ws + OFF_VTINB);
    unsigned short* VTB   = (unsigned short*)(ws + OFF_VTB);
    unsigned short* WBF   = (unsigned short*)(ws + OFF_WBF);
    unsigned short* HBB   = (unsigned short*)(ws + OFF_HB);
    float*          Cb    = ws + OFF_C;
    float*          ZI    = ws + OFF_ZI;
    float*          ZH    = ws + OFF_ZH;
    float*          OUTS  = ws + OFF_OUTS;
    float*          Qb    = ws + OFF_Q;
    float*          SCOR  = ws + OFF_SCORES;
    unsigned short* ATTB  = (unsigned short*)(ws + OFF_ATTB);
    float*          ATT2  = ws + OFF_ATT2;
    float*          ACC   = ws + OFF_ACC;
    unsigned short* OUTSB = (unsigned short*)(ws + OFF_OUTSB);
    float*          VAL   = ws + OFF_VAL;
    float*          KB    = ws + OFF_KB;
    unsigned short* WKB   = (unsigned short*)(ws + OFF_WKB);
    unsigned short* WVB   = (unsigned short*)(ws + OFF_WVB);
    float*          outv  = (float*)d_out;

    zero_kernel<<<1, 32, 0, stream>>>(ACC, 4);

    // ---- one-time per call: convert weights to bf16 ----
    struct { const float* src; unsigned short* dst; int n; } wcv[] = {
        { emb_cat_W, WBF + WB_EMB_CAT, 512*1024 }, { emb_sub_W, WBF + WB_EMB_SUB, 512*1024 },
        { emb_con_W, WBF + WB_EMB_CON, 512*1024 }, { f1_this_W, WBF + WB_F1_THIS, 512*512 },
        { f1_last_W, WBF + WB_F1_LAST, 512*512 },  { f2_last_W, WBF + WB_F2_LAST, 1024*1024 },
        { f2_this_W, WBF + WB_F2_THIS, 512*512 },  { out_cat_W, WBF + WB_OUT_CAT, CATN*512 },
        { out_sub_W, WBF + WB_OUT_SUB, SUBN*1024 },{ out_con_W, WBF + WB_OUT_CON, CONN*1536 },
        { fuse_W,    WBF + WB_FUSE,    512*1536 }, { mlp_W,     WBF + WB_MLP,     512*2048 },
        { W_ih,      WBF + WB_WIH,     1024*512 }, { W_hh,      WBF + WB_WHH,     1024*256 },
        { Wq,        WBF + WB_WQ,      256*256 },  { Wout,      WBF + WB_WOUT,    256*1024 },
        { Wk,        WKB,              256*256 },  { Wv,        WVB,              1024*256 },
    };
    for (auto& wc : wcv)
        cvt_bf16_kernel<<<(wc.n + 255) / 256, 256, 0, stream>>>(wc.src, wc.dst, wc.n);

    // ---- inputs -> bf16 operand buffers ----
    concat_emb_kernel<<<(ROWS * DD) / 256, 256, 0, stream>>>(visual, text, EMBB, VTINB);
    user_embed_kernel<<<(ROWS * 256) / 256, 256, 0, stream>>>(user, user_emb, VTINB);
    {   // user_des = relu(ufeat @ proj_u_W.T + b) -> VT_INbf [1280:1536)  (K=400: generic)
        dim3 g((256 + 31) / 32, (ROWS + 31) / 32), b(128);
        gemm_wmma<<<g, b, 0, stream>>>(user + 1, UDIM + 1, proj_u_W, UDIM, proj_u_b,
                                       nullptr, 0, VTINB + 1280, 2048, ROWS, 256, UDIM, 1);
    }

    // ---- category embeddings ----
    gemmb(stream, EMBB, 1024, WBF + WB_EMB_CAT, 1024, emb_cat_b,
          CATE, 512, CATEB, 512, ROWS, 512, 1024, 0);
    gemmb(stream, EMBB, 1024, WBF + WB_EMB_SUB, 1024, emb_sub_b,
          SUBF, 1024, SUBFB, 1024, ROWS, 512, 1024, 0);
    gemmb(stream, EMBB, 1024, WBF + WB_EMB_CON, 1024, emb_con_b,
          nullptr, 0, CONFB, 1536, ROWS, 512, 1024, 0);

    // ---- fusion 1 ----
    gemmb(stream, SUBFB, 1024, WBF + WB_F1_THIS, 512, f1_this_b,
          SCR, 512, nullptr, 0, ROWS, 512, 512, 0);
    gemmb(stream, CATEB, 512, WBF + WB_F1_LAST, 512, f1_last_b,
          SCR + 4194304ULL, 512, nullptr, 0, ROWS, 512, 512, 0);
    gate1_kernel<<<(ROWS * 512) / 256, 256, 0, stream>>>(SCR + 4194304ULL, SCR, CATE,
                                                         SUBF, SUBFB);

    // ---- fusion 2 ----
    gemmb(stream, SUBFB, 1024, WBF + WB_F2_LAST, 1024, f2_last_b,
          SCR, 1024, nullptr, 0, ROWS, 1024, 1024, 0);
    gemmb(stream, CONFB, 1536, WBF + WB_F2_THIS, 512, f2_this_b,
          SCR + 8388608ULL, 512, nullptr, 0, ROWS, 512, 512, 0);
    gate2_kernel<<<(ROWS * 1024) / 256, 256, 0, stream>>>(SCR, SCR + 8388608ULL, SUBF, CONFB);

    // ---- classification heads + CE ----
    gemmb(stream, CATEB, 512, WBF + WB_OUT_CAT, 512, out_cat_b,
          SCR, CATN, nullptr, 0, ROWS, CATN, 512, 1);
    ce_kernel<<<ROWS, 256, 0, stream>>>(SCR, CATN, CATN, category, 0, 0, ACC + 0);
    gemmb(stream, SUBFB, 1024, WBF + WB_OUT_SUB, 1024, out_sub_b,
          SCR, SUBN, nullptr, 0, ROWS, SUBN, 1024, 1);
    ce_kernel<<<ROWS, 256, 0, stream>>>(SCR, SUBN, SUBN, category, 1, 0, ACC + 1);
    for (int m0 = 0; m0 < ROWS; m0 += 2048) {
        gemmb(stream, CONFB + (size_t)m0 * 1536, 1536, WBF + WB_OUT_CON, 1536, out_con_b,
              SCR, CONN, nullptr, 0, 2048, CONN, 1536, 1);
        ce_kernel<<<2048, 256, 0, stream>>>(SCR, CONN, CONN, category, 2, m0, ACC + 2);
    }

    // ---- cat_feat -> VT_INbf [1536:2048); v_t = relu(mlp(VT_IN)) ----
    gemmb(stream, CONFB, 1536, WBF + WB_FUSE, 1536, fuse_b,
          nullptr, 0, VTINB + 1536, 2048, ROWS, 512, 1536, 1);
    gemmb(stream, VTINB, 2048, WBF + WB_MLP, 2048, mlp_b,
          nullptr, 0, VTB, 512, ROWS, 512, 2048, 1);

    // ---- LSTM ----
    cvt_bf16_kernel<<<(BB * HH) / 256, 256, 0, stream>>>(h0, HBB, BB * HH);
    hipMemcpyAsync(Cb, c0, (size_t)BB * HH * sizeof(float), hipMemcpyDeviceToDevice, stream);
    for (int t = 0; t < SS; ++t) {
        gemmb(stream, VTB + (size_t)t * 512, SS * 512, WBF + WB_WIH, 512, b_ih,
              ZI, 1024, nullptr, 0, BB, 1024, 512, 0);
        gemmb(stream, HBB, HH, WBF + WB_WHH, HH, b_hh,
              ZH, 1024, nullptr, 0, BB, 1024, HH, 0);
        lstm_cell_kernel<<<BB, 256, 0, stream>>>(ZI, ZH, HBB, Cb,
            OUTS + (size_t)t * BB * HH, OUTSB + (size_t)t * BB * HH,
            ln_ih_g, ln_ih_b, ln_hh_g, ln_hh_b, ln_c_g, ln_c_b);
    }

    // ---- attention: K and VAL via WMMA GEMMs ----
    gemmb(stream, HBB, HH, WBF + WB_WQ, HH, bq, Qb, DKK, nullptr, 0, BB, DKK, HH, 0);
    gemmb(stream, OUTSB, HH, WKB, HH, bk, KB, DKK, nullptr, 0, ROWS, DKK, HH, 0);
    gemmb(stream, OUTSB, HH, WVB, HH, nullptr, VAL, NHH * DKK, nullptr, 0,
          ROWS, NHH * DKK, HH, 0);
    attn_score_kernel<<<ROWS, 256, 0, stream>>>(KB, Qb, SCOR);
    attn_softmax_kernel<<<BB * NHH, 32, 0, stream>>>(SCOR);
    attn_reduce_kernel<<<(BB * NHH * DKK) / 256, 256, 0, stream>>>(VAL, SCOR, bv, ATTB);
    gemmb(stream, ATTB, NHH * DKK, WBF + WB_WOUT, NHH * DKK, bout,
          ATT2, DKK, nullptr, 0, BB, DKK, NHH * DKK, 0);

    // ---- final projection + losses -> d_out = [loss, output(512)] ----
    final_out_kernel<<<BB, 256, 0, stream>>>(ATT2, Wo, bo, label, outv + 1, ACC + 3);
    loss_kernel<<<1, 1, 0, stream>>>(ACC, outv);
}
